// GNNPhishingDetector_41987600285851
// MI455X (gfx1250) — compile-verified, matrix operand order checked by reading, not run on hardware
//
#include <hip/hip_runtime.h>
#include <hip/hip_bf16.h>

typedef float        v2f   __attribute__((ext_vector_type(2)));
typedef float        v8f   __attribute__((ext_vector_type(8)));
typedef __bf16       v4bf  __attribute__((ext_vector_type(4)));
typedef __bf16       v8bf  __attribute__((ext_vector_type(8)));
typedef __bf16       v16bf __attribute__((ext_vector_type(16)));
typedef unsigned int u32x4 __attribute__((ext_vector_type(4)));
typedef int          i32x8 __attribute__((ext_vector_type(8)));
typedef int          i32x4 __attribute__((ext_vector_type(4)));

#if __has_builtin(__builtin_amdgcn_tensor_load_to_lds) && __has_builtin(__builtin_amdgcn_s_wait_tensorcnt)
#define HAS_TDM 1
#else
#define HAS_TDM 0
#endif

// ---------------------------------------------------------------- zero
__global__ void k_zero(float* __restrict__ p, long n) {
  long i = (long)blockIdx.x * blockDim.x + threadIdx.x;
  long st = (long)gridDim.x * blockDim.x;
  for (; i < n; i += st) p[i] = 0.f;
}

// ---------------------------------------------------------------- edge pass 1 (4-dim features)
__global__ void k_edge1(const float* __restrict__ x, const long long* __restrict__ ei,
                        float* __restrict__ agg1, float* __restrict__ deg, int E) {
  int e = blockIdx.x * blockDim.x + threadIdx.x;
  if (e >= E) return;
  int s = (int)ei[e];
  int d = (int)ei[(long)E + e];
  const float4 xv = *(const float4*)(x + (long)s * 4);
  float* a = agg1 + (long)d * 4;
  atomicAdd(a + 0, xv.x);
  atomicAdd(a + 1, xv.y);
  atomicAdd(a + 2, xv.z);
  atomicAdd(a + 3, xv.w);
  atomicAdd(deg + d, 1.0f);
}

// ---------------------------------------------------------------- weight pre-pack into WMMA lane order
// pack2 (bf16): [mat][ct 8][kt 4][lane 32][j 16] ; (k,n) = (kt*32 + (lane>>4)*8 + (j<8 ? j : j+8), ct*16 + (lane&15))
// pack1 (f32) : [mat][ct 8][lane 32][j 2]        ; (k,n) = ((lane>>4)*2 + j, ct*16 + (lane&15))
__global__ void k_pack(const float* __restrict__ W1l, const float* __restrict__ W1r,
                       const float* __restrict__ W2l, const float* __restrict__ W2r,
                       float* __restrict__ pack1, __bf16* __restrict__ pack2) {
  int t = blockIdx.x * blockDim.x + threadIdx.x;
  if (t < 32768) {
    int j = t & 15, lane = (t >> 4) & 31, kt = (t >> 9) & 3, ct = (t >> 11) & 7, mat = (t >> 14) & 1;
    int n = lane & 15;
    int k = kt * 32 + ((lane >> 4) << 3) + (j < 8 ? j : j + 8);
    const float* W = mat ? W2r : W2l;
    pack2[t] = (__bf16)W[k * 128 + ct * 16 + n];
  } else if (t < 32768 + 1024) {
    int u = t - 32768;
    int j = u & 1, lane = (u >> 1) & 31, ct = (u >> 6) & 7, mat = (u >> 9) & 1;
    int n = lane & 15;
    int k = ((lane >> 4) << 1) + j;
    const float* W = mat ? W1r : W1l;
    pack1[u] = W[k * 128 + ct * 16 + n];
  }
}

// ---------------------------------------------------------------- layer 1: [N,4] x [4,128] via V_WMMA_F32_16X16X4_F32
// Output tiles staged in LDS, written back as coalesced b128 stores.
__global__ void k_layer1(const float* __restrict__ x, const float* __restrict__ agg1,
                         const float* __restrict__ deg, const float* __restrict__ b1,
                         const float* __restrict__ pack1, __bf16* __restrict__ h1, int N) {
  __shared__ __align__(32) __bf16 st[64 * 128];   // 16 KB staging for this block's 64 rows
  int tid = threadIdx.x;
  int lane = tid & 31, wave = tid >> 5;
  int base = blockIdx.x * 64 + wave * 16;
  int m = lane & 15, half = lane >> 4;

  if (base < N) {
    int row = base + m;
    int rowc = row < N ? row : N - 1;
    float rd = 1.f / fmaxf(deg[rowc], 1.f);
    v2f aM, aX;
#pragma unroll
    for (int j = 0; j < 2; ++j) {
      int k = half * 2 + j;
      aM[j] = agg1[(long)rowc * 4 + k] * rd;
      aX[j] = x[(long)rowc * 4 + k];
    }
    for (int ct = 0; ct < 8; ++ct) {
      v2f bl = *(const v2f*)(pack1 + (ct * 32 + lane) * 2);
      v2f br = *(const v2f*)(pack1 + 512 + (ct * 32 + lane) * 2);
      v8f c = {};
      c = __builtin_amdgcn_wmma_f32_16x16x4_f32(false, aM, false, bl, (short)0, c, false, false);
      c = __builtin_amdgcn_wmma_f32_16x16x4_f32(false, aX, false, br, (short)0, c, false, false);
      int col = ct * 16 + m;
      float bias = b1[col];
#pragma unroll
      for (int r = 0; r < 8; ++r) {
        float v = fmaxf(c[r] + bias, 0.f);
        st[(wave * 16 + r + half * 8) * 128 + col] = (__bf16)v;   // ds_store_b16
      }
    }
  }
  __syncthreads();

  int nrow = N - blockIdx.x * 64;
  if (nrow > 64) nrow = 64;
  for (int i = tid; i < 64 * 16; i += 128) {       // 16-byte chunks, coalesced
    int row = i >> 4;
    if (row < nrow) {
      int cc = (i & 15) * 8;
      *(v8bf*)(h1 + ((long)(blockIdx.x * 64 + row)) * 128 + cc) = *(const v8bf*)(st + row * 128 + cc);
    }
  }
}

// ---------------------------------------------------------------- edge pass 2: gather h1(bf16) -> agg2(f32) atomics
__global__ void k_edge2(const __bf16* __restrict__ h1, const long long* __restrict__ ei,
                        float* __restrict__ agg2, int E) {
  long t = (long)blockIdx.x * blockDim.x + threadIdx.x;
  long total = (long)E * 32;
  if (t >= total) return;
  int e = (int)(t >> 5);
  int c = (int)(t & 31);
  int s = (int)ei[e];
  int d = (int)ei[(long)E + e];
  v4bf hv = *(const v4bf*)(h1 + (long)s * 128 + c * 4);
  float* a = agg2 + (long)d * 128 + c * 4;
  atomicAdd(a + 0, (float)hv[0]);
  atomicAdd(a + 1, (float)hv[1]);
  atomicAdd(a + 2, (float)hv[2]);
  atomicAdd(a + 3, (float)hv[3]);
}

// ---------------------------------------------------------------- layer 2: [N,128]x[128,128]x2 via V_WMMA_F32_16X16X32_BF16
// B operands staged in LDS via the Tensor Data Mover (TDM); bias/ReLU/mean-pool fused.
__global__ void k_layer2(const __bf16* __restrict__ h1, const float* __restrict__ agg2,
                         const float* __restrict__ deg, const float* __restrict__ b2,
                         const __bf16* __restrict__ pack2, float* __restrict__ pooled, int N) {
  __shared__ float smp[128];
  __shared__ __align__(64) __bf16 smB[32768];      // 64 KB: both packed W2 matrices
  int tid = threadIdx.x;
  if (tid < 128) smp[tid] = 0.f;

#if HAS_TDM
  if ((tid >> 5) == 0) {
    // Build a minimal 2-D tensor descriptor (D#): 16384 dwords, tile = whole tensor.
    unsigned lds = (unsigned)(unsigned long long)(size_t)&smB[0];
    unsigned long long ga = (unsigned long long)(size_t)pack2;
    u32x4 g0 = {};
    g0[0] = 1u;                                            // count=1, user descriptor
    g0[1] = lds;                                           // lds_addr
    g0[2] = (unsigned)ga;                                  // global_addr[31:0]
    g0[3] = (unsigned)((ga >> 32) & 0x1FFFFFFu) | 0x80000000u;  // global_addr[56:32] | type=2
    i32x8 g1 = {};
    g1[0] = (int)(2u << 16);                               // data_size = 4B, no multicast
    g1[1] = (int)(16384u << 16);                           // tensor_dim0[15:0]  (bits 63:48)
    g1[2] = (int)(((16384u >> 16) & 0xFFFFu) | (1u << 16));// tensor_dim0[31:16] | tensor_dim1 = 1
    g1[3] = (int)(16384u << 16);                           // tile_dim0 = 16384  (bits 127:112)
    g1[4] = 1;                                             // tile_dim1 = 1
    g1[5] = 16384;                                         // tensor_dim0_stride lo32
    i32x4 g2 = {}, g3 = {};
    i32x8 g4 = {};
    __builtin_amdgcn_tensor_load_to_lds(g0, g1, g2, g3, g4, 0);  // 6-arg form (clang-23 lane)
    __builtin_amdgcn_s_wait_tensorcnt(0);
  }
#else
  {
    const uint4* src = (const uint4*)pack2;
    uint4* dst = (uint4*)smB;
    for (int i = tid; i < 4096; i += 128) dst[i] = src[i];
  }
#endif
  __syncthreads();

  int lane = tid & 31, wave = tid >> 5;
  int base = (blockIdx.x * 4 + wave) * 16;
  int m = lane & 15, half = lane >> 4;

  if (base < N) {
    int row = base + m;
    int rowc = row < N ? row : N - 1;
    float rd = 1.f / fmaxf(deg[rowc], 1.f);

    v16bf aH[4], aM[4];
#pragma unroll
    for (int kt = 0; kt < 4; ++kt) {
      int koff = kt * 32 + half * 8;                 // ISA 16-bit A layout: lanes16-31 shifted by +8
      v8bf lo = *(const v8bf*)(h1 + (long)rowc * 128 + koff);
      v8bf hi = *(const v8bf*)(h1 + (long)rowc * 128 + koff + 16);
      aH[kt] = __builtin_shufflevector(lo, hi, 0, 1, 2, 3, 4, 5, 6, 7, 8, 9, 10, 11, 12, 13, 14, 15);
      v8f flo = *(const v8f*)(agg2 + (long)rowc * 128 + koff);
      v8f fhi = *(const v8f*)(agg2 + (long)rowc * 128 + koff + 16);
      v16bf am;
#pragma unroll
      for (int j = 0; j < 8; ++j) {
        am[j]     = (__bf16)(flo[j] * rd);
        am[8 + j] = (__bf16)(fhi[j] * rd);
      }
      aM[kt] = am;
    }

    for (int ct = 0; ct < 8; ++ct) {
      v8f c = {};
#pragma unroll
      for (int kt = 0; kt < 4; ++kt) {
        v16bf bl = *(const v16bf*)(smB + ((ct * 4 + kt) * 32 + lane) * 16);          // ds_load_b128 x2
        v16bf br = *(const v16bf*)(smB + 16384 + ((ct * 4 + kt) * 32 + lane) * 16);
        c = __builtin_amdgcn_wmma_f32_16x16x32_bf16(false, aM[kt], false, bl, (short)0, c, false, false);
        c = __builtin_amdgcn_wmma_f32_16x16x32_bf16(false, aH[kt], false, br, (short)0, c, false, false);
      }
      int col = ct * 16 + m;
      float bias = b2[col];
      float s = 0.f;
#pragma unroll
      for (int r = 0; r < 8; ++r) {
        int rr = base + r + half * 8;
        float v = fmaxf(c[r] + bias, 0.f);
        if (rr < N) s += v;
      }
      atomicAdd(&smp[col], s);                       // ds_add_f32
    }
  }
  __syncthreads();
  if (tid < 128) atomicAdd(&pooled[tid], smp[tid]);
}

// ---------------------------------------------------------------- head: mean-pool scale + 3 dense layers + sigmoid
__global__ void k_head(const float* __restrict__ pooled, const float* __restrict__ Wp,
                       const float* __restrict__ bp, const float* __restrict__ Wc1,
                       const float* __restrict__ bc1, const float* __restrict__ Wc2,
                       const float* __restrict__ bc2, float* __restrict__ out, int N) {
  __shared__ float sp[128], se[256], sh[128];
  int t = threadIdx.x;
  if (t < 128) sp[t] = pooled[t] / (float)N;
  __syncthreads();
  {
    float acc = bp[t];
    for (int k = 0; k < 128; ++k) acc += sp[k] * Wp[k * 256 + t];
    se[t] = fmaxf(acc, 0.f);
  }
  __syncthreads();
  if (t < 128) {
    float acc = bc1[t];
    for (int k = 0; k < 256; ++k) acc += se[k] * Wc1[k * 128 + t];
    sh[t] = fmaxf(acc, 0.f);
  }
  __syncthreads();
  if (t == 0) {
    float acc = bc2[0];
    for (int k = 0; k < 128; ++k) acc += sh[k] * Wc2[k];
    out[0] = 1.f / (1.f + __expf(-acc));
  }
}

// ---------------------------------------------------------------- launch
extern "C" void kernel_launch(void* const* d_in, const int* in_sizes, int n_in,
                              void* d_out, int out_size, void* d_ws, size_t ws_size,
                              hipStream_t stream) {
  const float*     x   = (const float*)d_in[0];
  const long long* ei  = (const long long*)d_in[1];
  const float*     W1l = (const float*)d_in[2];
  const float*     W1r = (const float*)d_in[3];
  const float*     b1  = (const float*)d_in[4];
  const float*     W2l = (const float*)d_in[5];
  const float*     W2r = (const float*)d_in[6];
  const float*     b2  = (const float*)d_in[7];
  const float*     Wp  = (const float*)d_in[8];
  const float*     bp  = (const float*)d_in[9];
  const float*     Wc1 = (const float*)d_in[10];
  const float*     bc1 = (const float*)d_in[11];
  const float*     Wc2 = (const float*)d_in[12];
  const float*     bc2 = (const float*)d_in[13];

  const int N = in_sizes[0] / 4;
  const int E = in_sizes[1] / 2;

  char* ws = (char*)d_ws;
  size_t off = 0;
  auto take = [&](size_t bytes) -> size_t {
    size_t o = off;
    off = (off + bytes + 255) & ~(size_t)255;
    return o;
  };
  // Regions needing zeroing first and contiguous:
  size_t o_agg2  = take((size_t)N * 128 * 4);   // f32 [N,128]
  size_t o_agg1  = take((size_t)N * 4 * 4);     // f32 [N,4]
  size_t o_deg   = take((size_t)N * 4);         // f32 [N]
  size_t o_pool  = take(128 * 4);               // f32 [128]
  size_t zero_end = off;
  size_t o_h1    = take((size_t)N * 128 * 2);   // bf16 [N,128]
  size_t o_pack1 = take(1024 * 4);              // f32 packed W1
  size_t o_pack2 = take(32768 * 2);             // bf16 packed W2

  float*  agg2   = (float*)(ws + o_agg2);
  float*  agg1   = (float*)(ws + o_agg1);
  float*  deg    = (float*)(ws + o_deg);
  float*  pooled = (float*)(ws + o_pool);
  __bf16* h1     = (__bf16*)(ws + o_h1);
  float*  pack1  = (float*)(ws + o_pack1);
  __bf16* pack2  = (__bf16*)(ws + o_pack2);

  // 1) zero accumulators (whole span [0, zero_end))
  long zwords = (long)(zero_end / 4);
  k_zero<<<4096, 256, 0, stream>>>((float*)ws, zwords);

  // 2) edge pass 1 + 3) weight pre-pack
  k_edge1<<<(E + 255) / 256, 256, 0, stream>>>(x, ei, agg1, deg, E);
  k_pack<<<(32768 + 1024 + 255) / 256, 256, 0, stream>>>(W1l, W1r, W2l, W2r, pack1, pack2);

  // 4) layer 1 (WMMA f32 16x16x4, LDS-staged coalesced stores)
  int nblk = (N + 63) / 64;
  k_layer1<<<nblk, 128, 0, stream>>>(x, agg1, deg, b1, pack1, h1, N);

  // 5) edge pass 2 (one wave per edge)
  long t2 = (long)E * 32;
  k_edge2<<<(int)((t2 + 255) / 256), 256, 0, stream>>>(h1, ei, agg2, E);

  // 6) layer 2 (WMMA bf16 16x16x32, TDM-staged B operands) + pooled reduction
  k_layer2<<<nblk, 128, 0, stream>>>(h1, agg2, deg, b2, pack2, pooled, N);

  // 7) head
  k_head<<<1, 256, 0, stream>>>(pooled, Wp, bp, Wc1, bc1, Wc2, bc2, (float*)d_out, N);
}
